// GPT_76605036692158
// MI455X (gfx1250) — matrix operand order repaired and checked by reading, not compiled, not run on hardware
//
#include <hip/hip_runtime.h>
#include <hip/hip_bf16.h>

#define B_    4
#define T_    1024
#define E_    768
#define H_    12
#define D_    64
#define L_    12
#define FF_   3072
#define MTOK  4096

typedef __attribute__((ext_vector_type(16))) __bf16 v16bf;
typedef __attribute__((ext_vector_type(8)))  float  v8f;
typedef __attribute__((ext_vector_type(4)))  unsigned u32x4;
typedef __attribute__((ext_vector_type(8)))  int      i32x8;
typedef __attribute__((ext_vector_type(4)))  int      i32x4;

#if defined(__has_builtin)
#  if __has_builtin(__builtin_amdgcn_tensor_load_to_lds) && \
      __has_builtin(__builtin_amdgcn_s_wait_tensorcnt)
#    define HAVE_TDM 1
#  endif
#endif
#ifndef HAVE_TDM
#  define HAVE_TDM 0
#endif

union FragBF { unsigned u[8]; v16bf v; };

__device__ __forceinline__ unsigned short f2bf(float f) {
  __bf16 b = (__bf16)f;                       // hardware cvt (RNE)
  return __builtin_bit_cast(unsigned short, b);
}
__device__ __forceinline__ float bf2f(unsigned short s) {
  return __uint_as_float(((unsigned)s) << 16);
}

// ---------------------------------------------------------------- embedding
__global__ __launch_bounds__(256)
void embed_kernel(const float* __restrict__ x, const float* __restrict__ Wte,
                  const float* __restrict__ bte, const float* __restrict__ wpe,
                  float* __restrict__ h) {
  int idx = blockIdx.x * 256 + threadIdx.x;       // over MTOK*E_
  int tok = idx / E_;
  int e   = idx - tok * E_;
  int t   = tok & (T_ - 1);
  h[idx] = x[tok] * Wte[e] + bte[e] + wpe[t * E_ + e];
}

// ---------------------------------------------------------------- layernorm
__global__ __launch_bounds__(256)
void layernorm_kernel(const float* __restrict__ h, const float* __restrict__ s,
                      const float* __restrict__ b, unsigned short* __restrict__ out,
                      float eps) {
  __shared__ float red[256];
  int row = blockIdx.x;
  int tid = threadIdx.x;
  const float* hp = h + (size_t)row * E_;
  float v0 = hp[tid], v1 = hp[tid + 256], v2 = hp[tid + 512];
  red[tid] = v0 + v1 + v2;
  __syncthreads();
  for (int off = 128; off > 0; off >>= 1) {
    if (tid < off) red[tid] += red[tid + off];
    __syncthreads();
  }
  float mu = red[0] * (1.0f / E_);
  __syncthreads();
  float d0 = v0 - mu, d1 = v1 - mu, d2 = v2 - mu;
  red[tid] = d0 * d0 + d1 * d1 + d2 * d2;
  __syncthreads();
  for (int off = 128; off > 0; off >>= 1) {
    if (tid < off) red[tid] += red[tid + off];
    __syncthreads();
  }
  float rstd = rsqrtf(red[0] * (1.0f / E_) + eps);
  unsigned short* op = out + (size_t)row * E_;
  op[tid]       = f2bf(d0 * rstd * s[tid]       + b[tid]);
  op[tid + 256] = f2bf(d1 * rstd * s[tid + 256] + b[tid + 256]);
  op[tid + 512] = f2bf(d2 * rstd * s[tid + 512] + b[tid + 512]);
}

// ---------------------------------------------------------------- GEMM (bf16 WMMA)
// C[M,N] = A[M,K](bf16) * Bw[K,N](fp32->bf16) + bias; optional GELU; optional
// residual-accumulate into fp32 outF (stride N). Block tile 128x128x64, 8 waves.
// Software-pipelined: next B tile fetched to registers before compute; A tile
// double-buffered in LDS and staged by the Tensor Data Mover when available.
#define BM  128
#define BN  128
#define BK2 64
#define LP2 66   // 66 halfwords = 33 dwords row stride, conflict-free + TDM pad

__global__ __launch_bounds__(256)
void gemm_bf16_kernel(const unsigned short* __restrict__ A, int lda,
                      const float* __restrict__ Bw,
                      const float* __restrict__ bias,
                      unsigned short* __restrict__ outB, int ldo,
                      float* __restrict__ outF,
                      int M, int N, int K, int act, int resid) {
  __shared__ unsigned short As[2][BM][LP2];   // double-buffered (TDM target)
  __shared__ unsigned short Bt[BN][LP2];      // transposed: [n][k], k contiguous
  int tid  = threadIdx.x;
  int lane = tid & 31;
  int w    = tid >> 5;
  int half = (lane >> 4) & 1;
  int l15  = lane & 15;
  int bm = blockIdx.y * BM;
  int bn = blockIdx.x * BN;
  int wm = (w & 3) * 32;      // wave tile: 32 rows x 64 cols
  int wn = (w >> 2) * 64;

  v8f zero = {0.f,0.f,0.f,0.f,0.f,0.f,0.f,0.f};
  v8f acc[2][4];
  for (int i = 0; i < 2; i++)
    for (int j = 0; j < 4; j++) acc[i][j] = zero;

  float4 breg[8];         // staged B tile (64x128 fp32)
#if !HAVE_TDM
  uint4  areg[4];         // staged A tile (128x64 bf16)
#endif

  auto issueA = [&](int k0, int buf) {
#if HAVE_TDM
    if (w == 0) {
      // D# group0: count=1 | lds_addr | global_addr | type=2
      unsigned long long ga =
          (unsigned long long)(size_t)(A + (size_t)bm * lda + k0);
      unsigned lba = (unsigned)(size_t)(&As[buf][0][0]);
      u32x4 g0;
      g0.x = 1u;
      g0.y = lba;
      g0.z = (unsigned)(ga & 0xFFFFFFFFu);
      g0.w = (unsigned)((ga >> 32) & 0x01FFFFFFu) | (2u << 30);
      // D# group1: data_size=2B, pad 1 dword every 32 dwords (row stride 66 hw)
      i32x8 g1;
      g1[0] = (int)((1u << 16) | (1u << 20) | (4u << 22));
      g1[1] = (int)((unsigned)lda << 16);            // tensor_dim0[15:0]
      g1[2] = (int)((unsigned)M << 16);              // dim0 hi=0 | tensor_dim1 lo
      g1[3] = (int)((unsigned)BK2 << 16);            // dim1 hi=0 | tile_dim0
      g1[4] = (int)BM;                               // tile_dim1 | tile_dim2=0
      g1[5] = (int)lda;                              // tensor_dim0_stride lo
      g1[6] = 0;
      g1[7] = 0;
      i32x4 g2 = {0, 0, 0, 0};
      i32x4 g3 = {0, 0, 0, 0};
#if __clang_major__ >= 23
      i32x8 g4 = {0, 0, 0, 0, 0, 0, 0, 0};
      __builtin_amdgcn_tensor_load_to_lds(g0, g1, g2, g3, g4, 0);
#else
      __builtin_amdgcn_tensor_load_to_lds(g0, g1, g2, g3, 0);
#endif
    }
#else
    (void)buf;
    #pragma unroll
    for (int it = 0; it < 4; it++) {
      int i   = tid + it * 256;
      int row = i >> 3;
      int c   = (i & 7) * 8;
      areg[it] = *(const uint4*)(A + (size_t)(bm + row) * lda + k0 + c);
    }
#endif
  };
#if !HAVE_TDM
  auto storeA = [&](int buf) {
    #pragma unroll
    for (int it = 0; it < 4; it++) {
      int i   = tid + it * 256;
      int row = i >> 3;
      int c   = (i & 7) * 8;
      unsigned short* dst = &As[buf][row][c];
      *(unsigned*)(dst + 0) = areg[it].x; *(unsigned*)(dst + 2) = areg[it].y;
      *(unsigned*)(dst + 4) = areg[it].z; *(unsigned*)(dst + 6) = areg[it].w;
    }
  };
#endif
  auto fetchB = [&](int k0) {
    #pragma unroll
    for (int it = 0; it < 8; it++) {
      int i  = tid + it * 256;
      int kk = i >> 5;
      int n4 = (i & 31) * 4;
      breg[it] = *(const float4*)(Bw + (size_t)(k0 + kk) * N + bn + n4);
    }
  };
  auto storeB = [&]() {
    #pragma unroll
    for (int it = 0; it < 8; it++) {
      int i  = tid + it * 256;
      int kk = i >> 5;
      int n4 = (i & 31) * 4;
      Bt[n4 + 0][kk] = f2bf(breg[it].x);
      Bt[n4 + 1][kk] = f2bf(breg[it].y);
      Bt[n4 + 2][kk] = f2bf(breg[it].z);
      Bt[n4 + 3][kk] = f2bf(breg[it].w);
    }
  };

  // -------- prologue: stage tile 0
  issueA(0, 0);
  fetchB(0);
#if !HAVE_TDM
  storeA(0);
#endif
  storeB();
#if HAVE_TDM
  if (w == 0) __builtin_amdgcn_s_wait_tensorcnt(0);
#endif
  __syncthreads();

  int nk = K / BK2;
  for (int kt = 0; kt < nk; kt++) {
    int  cur  = kt & 1;
    int  nxt  = cur ^ 1;
    bool more = (kt + 1) < nk;
    if (more) {
      issueA((kt + 1) * BK2, nxt);   // TDM -> As[nxt] (async) or global->regs
      fetchB((kt + 1) * BK2);        // global->regs, latency hidden by WMMAs
    }

    #pragma unroll
    for (int ks = 0; ks < BK2; ks += 32) {
      // A fragments for both m-tiles (ISA 16-bit A 16x32 layout)
      FragBF af[2];
      int kb = half * 8;
      #pragma unroll
      for (int mi = 0; mi < 2; mi++) {
        const unsigned short* rp = &As[cur][wm + mi * 16 + l15][ks];
        af[mi].u[0] = *(const unsigned*)(rp + kb + 0);
        af[mi].u[1] = *(const unsigned*)(rp + kb + 2);
        af[mi].u[2] = *(const unsigned*)(rp + kb + 4);
        af[mi].u[3] = *(const unsigned*)(rp + kb + 6);
        af[mi].u[4] = *(const unsigned*)(rp + 16 + kb + 0);
        af[mi].u[5] = *(const unsigned*)(rp + 16 + kb + 2);
        af[mi].u[6] = *(const unsigned*)(rp + 16 + kb + 4);
        af[mi].u[7] = *(const unsigned*)(rp + 16 + kb + 6);
      }
      #pragma unroll
      for (int ni = 0; ni < 4; ni++) {
        FragBF bb;
        const unsigned short* bp = &Bt[wn + ni * 16 + l15][ks + half * 8];
        #pragma unroll
        for (int v = 0; v < 8; v++) bb.u[v] = *(const unsigned*)(bp + 2 * v + (v >= 4 ? 8 : 0));
        acc[0][ni] = __builtin_amdgcn_wmma_f32_16x16x32_bf16(
            false, af[0].v, false, bb.v, (short)0, acc[0][ni], false, false);
        acc[1][ni] = __builtin_amdgcn_wmma_f32_16x16x32_bf16(
            false, af[1].v, false, bb.v, (short)0, acc[1][ni], false, false);
      }
    }

    __syncthreads();                 // all waves done reading Bt / As[cur]
    if (more) {
#if !HAVE_TDM
      storeA(nxt);
#endif
      storeB();                      // refill Bt for next stage
#if HAVE_TDM
      if (w == 0) __builtin_amdgcn_s_wait_tensorcnt(0);
#endif
    }
    __syncthreads();                 // publish next stage
  }

  // epilogue: C layout M = r + 8*half, N = l15 within each 16x16 tile
  #pragma unroll
  for (int mi = 0; mi < 2; mi++) {
    #pragma unroll
    for (int ni = 0; ni < 4; ni++) {
      int col = bn + wn + ni * 16 + l15;
      float bv = bias[col];
      #pragma unroll
      for (int r = 0; r < 8; r++) {
        int row = bm + wm + mi * 16 + r + 8 * half;
        float v = acc[mi][ni][r] + bv;
        if (act == 1) {  // tanh GELU
          float c3 = v * v * v;
          v = 0.5f * v * (1.0f + tanhf(0.7978845608028654f * (v + 0.044715f * c3)));
        }
        if (resid) outF[(size_t)row * N + col] += v;
        else       outB[(size_t)row * ldo + col] = f2bf(v);
      }
    }
  }
}

// ---------------------------------------------------------------- flash attention
// grid (T/128, H, B); block 256 = 8 waves; each wave owns 16 query rows.
__global__ __launch_bounds__(256)
void attention_kernel(const unsigned short* __restrict__ qkv,
                      unsigned short* __restrict__ y) {
  __shared__ unsigned short Qs[128][66];
  __shared__ unsigned short Ks[64][66];
  __shared__ unsigned short Vt[64][66];   // transposed: [d][key]
  __shared__ unsigned short Ps[128][66];
  int tid  = threadIdx.x;
  int lane = tid & 31;
  int w    = tid >> 5;
  int half = (lane >> 4) & 1;
  int l15  = lane & 15;
  int qb = blockIdx.x * 128;
  int hh = blockIdx.y;
  int bb_ = blockIdx.z;
  size_t tokbase = (size_t)bb_ * T_;
  const unsigned short* qptr = qkv + hh * D_;
  const unsigned short* kptr = qkv + E_ + hh * D_;
  const unsigned short* vptr = qkv + 2 * E_ + hh * D_;

  // Q tile 128x64
  for (int i = tid; i < 128 * 8; i += 256) {
    int row = i >> 3; int c = (i & 7) * 8;
    uint4 t4 = *(const uint4*)(qptr + (tokbase + qb + row) * (3 * E_) + c);
    unsigned short* dst = &Qs[row][c];
    *(unsigned*)(dst + 0) = t4.x; *(unsigned*)(dst + 2) = t4.y;
    *(unsigned*)(dst + 4) = t4.z; *(unsigned*)(dst + 6) = t4.w;
  }

  v8f zero = {0.f,0.f,0.f,0.f,0.f,0.f,0.f,0.f};
  float mrow[8], lrow[8];
  v8f o[4];
  #pragma unroll
  for (int r = 0; r < 8; r++) { mrow[r] = -__builtin_inff(); lrow[r] = 0.f; }
  #pragma unroll
  for (int dt = 0; dt < 4; dt++) o[dt] = zero;

  int kbq = half * 8;
  int nch = qb / 64 + 2;
  for (int kc = 0; kc < nch; kc++) {
    __syncthreads();
    int kb0 = kc * 64;
    // K chunk 64x64
    for (int i = tid; i < 64 * 8; i += 256) {
      int row = i >> 3; int c = (i & 7) * 8;
      uint4 t4 = *(const uint4*)(kptr + (tokbase + kb0 + row) * (3 * E_) + c);
      unsigned short* dst = &Ks[row][c];
      *(unsigned*)(dst + 0) = t4.x; *(unsigned*)(dst + 2) = t4.y;
      *(unsigned*)(dst + 4) = t4.z; *(unsigned*)(dst + 6) = t4.w;
    }
    // V chunk 64x64 -> transposed
    for (int i = tid; i < 64 * 32; i += 256) {
      int key = i >> 5; int d2 = (i & 31) * 2;
      unsigned uv = *(const unsigned*)(vptr + (tokbase + kb0 + key) * (3 * E_) + d2);
      Vt[d2][key]     = (unsigned short)(uv & 0xFFFFu);
      Vt[d2 + 1][key] = (unsigned short)(uv >> 16);
    }
    __syncthreads();

    // S = (1/8) * Q Kt over this wave's 16 rows
    v8f s[4];
    #pragma unroll
    for (int nt = 0; nt < 4; nt++) s[nt] = zero;
    #pragma unroll
    for (int ks = 0; ks < 64; ks += 32) {
      FragBF a;
      const unsigned short* rp = &Qs[w * 16 + l15][ks];
      a.u[0] = *(const unsigned*)(rp + kbq + 0);
      a.u[1] = *(const unsigned*)(rp + kbq + 2);
      a.u[2] = *(const unsigned*)(rp + kbq + 4);
      a.u[3] = *(const unsigned*)(rp + kbq + 6);
      a.u[4] = *(const unsigned*)(rp + 16 + kbq + 0);
      a.u[5] = *(const unsigned*)(rp + 16 + kbq + 2);
      a.u[6] = *(const unsigned*)(rp + 16 + kbq + 4);
      a.u[7] = *(const unsigned*)(rp + 16 + kbq + 6);
      #pragma unroll
      for (int nt = 0; nt < 4; nt++) {
        FragBF kb_;
        const unsigned short* bp = &Ks[nt * 16 + l15][ks + half * 8];
        #pragma unroll
        for (int v = 0; v < 8; v++) kb_.u[v] = *(const unsigned*)(bp + 2 * v + (v >= 4 ? 8 : 0));
        s[nt] = __builtin_amdgcn_wmma_f32_16x16x32_bf16(
            false, a.v, false, kb_.v, (short)0, s[nt], false, false);
      }
    }

    // streaming softmax (per q-row; stats live per (half, r))
    #pragma unroll
    for (int r = 0; r < 8; r++) {
      int q = qb + w * 16 + r + 8 * half;
      float sv[4];
      float rmax = -__builtin_inff();
      #pragma unroll
      for (int nt = 0; nt < 4; nt++) {
        float xv = s[nt][r] * 0.125f;
        int key = kb0 + nt * 16 + l15;
        if (key > q) xv = -__builtin_inff();
        sv[nt] = xv;
        rmax = fmaxf(rmax, xv);
      }
      for (int m = 1; m < 16; m <<= 1) rmax = fmaxf(rmax, __shfl_xor(rmax, m, 32));
      float mnew  = fmaxf(mrow[r], rmax);
      float scale = __expf(mrow[r] - mnew);
      mrow[r] = mnew;
      float psum = 0.f;
      #pragma unroll
      for (int nt = 0; nt < 4; nt++) {
        float p = __expf(sv[nt] - mnew);
        psum += p;
        Ps[w * 16 + r + 8 * half][nt * 16 + l15] = f2bf(p);
      }
      for (int m = 1; m < 16; m <<= 1) psum += __shfl_xor(psum, m, 32);
      lrow[r] = lrow[r] * scale + psum;
      #pragma unroll
      for (int dt = 0; dt < 4; dt++) o[dt][r] *= scale;
    }

    asm volatile("s_wait_dscnt 0" ::: "memory");  // P stores -> P loads (same wave)

    // O += P * V
    #pragma unroll
    for (int ks = 0; ks < 64; ks += 32) {
      FragBF a;
      const unsigned short* rp = &Ps[w * 16 + l15][ks];
      a.u[0] = *(const unsigned*)(rp + kbq + 0);
      a.u[1] = *(const unsigned*)(rp + kbq + 2);
      a.u[2] = *(const unsigned*)(rp + kbq + 4);
      a.u[3] = *(const unsigned*)(rp + kbq + 6);
      a.u[4] = *(const unsigned*)(rp + 16 + kbq + 0);
      a.u[5] = *(const unsigned*)(rp + 16 + kbq + 2);
      a.u[6] = *(const unsigned*)(rp + 16 + kbq + 4);
      a.u[7] = *(const unsigned*)(rp + 16 + kbq + 6);
      #pragma unroll
      for (int dt = 0; dt < 4; dt++) {
        FragBF vb;
        const unsigned short* bp = &Vt[dt * 16 + l15][ks + half * 8];
        #pragma unroll
        for (int v = 0; v < 8; v++) vb.u[v] = *(const unsigned*)(bp + 2 * v + (v >= 4 ? 8 : 0));
        o[dt] = __builtin_amdgcn_wmma_f32_16x16x32_bf16(
            false, a.v, false, vb.v, (short)0, o[dt], false, false);
      }
    }
  }

  // write y[tok][h*64 + d]
  #pragma unroll
  for (int dt = 0; dt < 4; dt++) {
    #pragma unroll
    for (int r = 0; r < 8; r++) {
      float ov = o[dt][r] / lrow[r];
      size_t tok = tokbase + qb + w * 16 + r + 8 * half;
      y[tok * E_ + hh * D_ + dt * 16 + l15] = f2bf(ov);
    }
  }
}

// ---------------------------------------------------------------- final GEMV (E -> 1)
__global__ __launch_bounds__(256)
void out_proj_kernel(const unsigned short* __restrict__ xb,
                     const float* __restrict__ Wout,
                     const float* __restrict__ bout,
                     float* __restrict__ out) {
  __shared__ float red[256];
  int row = blockIdx.x;
  int tid = threadIdx.x;
  const unsigned short* xp = xb + (size_t)row * E_;
  red[tid] = bf2f(xp[tid]) * Wout[tid]
           + bf2f(xp[tid + 256]) * Wout[tid + 256]
           + bf2f(xp[tid + 512]) * Wout[tid + 512];
  __syncthreads();
  for (int off = 128; off > 0; off >>= 1) {
    if (tid < off) red[tid] += red[tid + off];
    __syncthreads();
  }
  if (tid == 0) out[row] = red[0] + bout[0];
}

// ---------------------------------------------------------------- launch
extern "C" void kernel_launch(void* const* d_in, const int* in_sizes, int n_in,
                              void* d_out, int out_size, void* d_ws, size_t ws_size,
                              hipStream_t stream) {
  const float* x    = (const float*)d_in[0];
  const float* Wte  = (const float*)d_in[1];
  const float* bte  = (const float*)d_in[2];
  const float* wpe  = (const float*)d_in[3];
  const float* ln1s = (const float*)d_in[4];
  const float* ln1b = (const float*)d_in[5];
  const float* Wqkv = (const float*)d_in[6];
  const float* bqkv = (const float*)d_in[7];
  const float* Wo   = (const float*)d_in[8];
  const float* bo   = (const float*)d_in[9];
  const float* ln2s = (const float*)d_in[10];
  const float* ln2b = (const float*)d_in[11];
  const float* Wfc  = (const float*)d_in[12];
  const float* bfc  = (const float*)d_in[13];
  const float* Wp2  = (const float*)d_in[14];
  const float* bp2  = (const float*)d_in[15];
  const float* lnfs = (const float*)d_in[16];
  const float* lnfb = (const float*)d_in[17];
  const float* Wout = (const float*)d_in[18];
  const float* bout = (const float*)d_in[19];

  char* ws = (char*)d_ws;
  size_t off = 0;
  auto alloc = [&](size_t bytes) {
    void* p = ws + off;
    off = (off + bytes + 255) & ~(size_t)255;
    return p;
  };
  float*          h    = (float*)alloc((size_t)MTOK * E_ * 4);            // fp32 residual
  unsigned short* xb   = (unsigned short*)alloc((size_t)MTOK * E_ * 2);   // LN out (bf16)
  unsigned short* qkvb = (unsigned short*)alloc((size_t)MTOK * 3 * E_ * 2);
  unsigned short* yb   = (unsigned short*)alloc((size_t)MTOK * E_ * 2);
  unsigned short* fcb  = (unsigned short*)alloc((size_t)MTOK * FF_ * 2);

  embed_kernel<<<(MTOK * E_) / 256, 256, 0, stream>>>(x, Wte, bte, wpe, h);

  for (int l = 0; l < L_; l++) {
    layernorm_kernel<<<MTOK, 256, 0, stream>>>(h, ln1s + l * E_, ln1b + l * E_, xb, 1e-5f);
    gemm_bf16_kernel<<<dim3((3 * E_) / 128, MTOK / 128), 256, 0, stream>>>(
        xb, E_, Wqkv + (size_t)l * E_ * 3 * E_, bqkv + (size_t)l * 3 * E_,
        qkvb, 3 * E_, nullptr, MTOK, 3 * E_, E_, 0, 0);
    attention_kernel<<<dim3(T_ / 128, H_, B_), 256, 0, stream>>>(qkvb, yb);
    gemm_bf16_kernel<<<dim3(E_ / 128, MTOK / 128), 256, 0, stream>>>(
        yb, E_, Wo + (size_t)l * E_ * E_, bo + (size_t)l * E_,
        nullptr, 0, h, MTOK, E_, E_, 0, 1);
    layernorm_kernel<<<MTOK, 256, 0, stream>>>(h, ln2s + l * E_, ln2b + l * E_, xb, 1e-5f);
    gemm_bf16_kernel<<<dim3(FF_ / 128, MTOK / 128), 256, 0, stream>>>(
        xb, E_, Wfc + (size_t)l * E_ * FF_, bfc + (size_t)l * FF_,
        fcb, FF_, nullptr, MTOK, FF_, E_, 1, 0);
    gemm_bf16_kernel<<<dim3(E_ / 128, MTOK / 128), 256, 0, stream>>>(
        fcb, FF_, Wp2 + (size_t)l * FF_ * E_, bp2 + (size_t)l * E_,
        nullptr, 0, h, MTOK, E_, FF_, 0, 1);
  }

  layernorm_kernel<<<MTOK, 256, 0, stream>>>(h, lnfs, lnfb, xb, 1e-6f);
  out_proj_kernel<<<MTOK, 256, 0, stream>>>(xb, Wout, bout, (float*)d_out);
}
